// MultiHeadAttention_16192026706167
// MI455X (gfx1250) — matrix operand order repaired and checked by reading, not compiled, not run on hardware
//
#include <hip/hip_runtime.h>

#define EMB   1024
#define SEQ   2048
#define BATCH 4
#define HEADS 16
#define HDIM  64
#define E3    (3 * EMB)

typedef __attribute__((ext_vector_type(16))) _Float16 v16h;
typedef __attribute__((ext_vector_type(8)))  float    v8f;

union Frag {
    uint4 u4[2];
    v16h  h;
};

static __device__ __forceinline__ v8f wmma16(v16h a, v16h b, v8f c) {
    // D = A(16x32 f16) x B(32x16 f16) + C(16x16 f32)
    return __builtin_amdgcn_wmma_f32_16x16x32_f16(
        /*neg_a=*/false, a, /*neg_b=*/false, b,
        /*c_mod=*/(short)0, c, /*reuse_a=*/false, /*reuse_b=*/false);
}

// async global -> LDS copy of 16 bytes per lane (ASYNCcnt-tracked DMA path).
// Low 32 bits of a generic LDS pointer == workgroup-relative LDS offset.
static __device__ __forceinline__ void async_copy16(const _Float16* gptr,
                                                    _Float16* lptr) {
    unsigned lds = (unsigned)(size_t)lptr;
    unsigned long long ga = (unsigned long long)(size_t)gptr;
    asm volatile("global_load_async_to_lds_b128 %0, %1, off"
                 :: "v"(lds), "v"(ga) : "memory");
}
static __device__ __forceinline__ void wait_async0() {
    asm volatile("s_wait_asynccnt 0x0" ::: "memory");
}

// ---------------------------------------------------------------- convert
__global__ void cvt_f32_to_f16(const float* __restrict__ in,
                               _Float16* __restrict__ out, int n) {
    int i = blockIdx.x * blockDim.x + threadIdx.x;
    int stride = gridDim.x * blockDim.x;
    for (; i < n; i += stride) out[i] = (_Float16)in[i];
}

// ---------------------------------------------------------------- GEMM
// C[M,N] = A[M,K] @ W[K,N] + bias ; f16 in, f32 accumulate.
// Block tile 128x128, K-step 32, double-buffered LDS:
//   A tile via async global->LDS DMA, W tile via reg-staged transpose.
// 8 waves, each 64(M)x32(N): 8 WMMAs per 6 fragment loads per K-step.
#define APAD 16   // row padding (32B): keeps 16B align, spreads banks

template <bool OUT_F32>
__global__ __launch_bounds__(256) void gemm_f16(
    const _Float16* __restrict__ A,
    const _Float16* __restrict__ W,
    const float* __restrict__ bias,
    void* __restrict__ Cout,
    int M, int N, int K)
{
    __shared__ alignas(16) _Float16 As[2][128][32 + APAD];  // [buf][m][k]
    __shared__ alignas(16) _Float16 Wt[2][128][32 + APAD];  // [buf][n][k]

    const int t    = threadIdx.x;
    const int lane = t & 31;
    const int wave = t >> 5;
    const int hi   = (lane >> 4) & 1;
    const int l16  = lane & 15;
    const int wm   = wave & 1;     // 2 M-waves -> 64 rows each
    const int wn   = wave >> 1;    // 4 N-waves -> 32 cols each
    const int m0   = blockIdx.y * 128;
    const int n0   = blockIdx.x * 128;
    const int nk   = K >> 5;

    uint4 wreg[2];

    auto issueA = [&](int k0, int buf) {   // 128x32: 2 b128 chunks / thread
        #pragma unroll
        for (int i = 0; i < 2; ++i) {
            int c   = t + i * 256;
            int row = c >> 2;
            int col = (c & 3) * 8;
            async_copy16(&A[(size_t)(m0 + row) * K + k0 + col],
                         &As[buf][row][col]);
        }
    };
    auto loadW = [&](int k0) {             // 32x128 row-major into regs
        #pragma unroll
        for (int i = 0; i < 2; ++i) {
            int c    = t + i * 256;
            int krow = c >> 4;
            int ncol = (c & 15) * 8;
            wreg[i] = *(const uint4*)&W[(size_t)(k0 + krow) * N + n0 + ncol];
        }
    };
    auto storeW = [&](int buf) {           // scatter transposed: Wt[n][k]
        #pragma unroll
        for (int i = 0; i < 2; ++i) {
            int c    = t + i * 256;
            int krow = c >> 4;
            int ncol = (c & 15) * 8;
            _Float16 tmp[8];
            *(uint4*)tmp = wreg[i];
            #pragma unroll
            for (int e = 0; e < 8; ++e) Wt[buf][ncol + e][krow] = tmp[e];
        }
    };

    // prologue: stage tile 0
    issueA(0, 0);
    loadW(0);
    storeW(0);
    wait_async0();
    __syncthreads();

    v8f acc[4][2] = {};

    for (int kt = 0; kt < nk; ++kt) {
        const int  cur  = kt & 1;
        const bool more = (kt + 1) < nk;
        if (more) {                         // overlap next-tile fetch
            issueA((kt + 1) * 32, cur ^ 1);
            loadW((kt + 1) * 32);
        }

        Frag fa[4], fb[2];
        #pragma unroll
        for (int im = 0; im < 4; ++im) {
            int mr = wm * 64 + im * 16 + l16;
            fa[im].u4[0] = *(const uint4*)&As[cur][mr][hi * 8];
            fa[im].u4[1] = *(const uint4*)&As[cur][mr][16 + hi * 8];
        }
        #pragma unroll
        for (int in_ = 0; in_ < 2; ++in_) {
            int nr = wn * 32 + in_ * 16 + l16;
            fb[in_].u4[0] = *(const uint4*)&Wt[cur][nr][hi * 8];
            fb[in_].u4[1] = *(const uint4*)&Wt[cur][nr][16 + hi * 8];
        }
        #pragma unroll
        for (int im = 0; im < 4; ++im)
            #pragma unroll
            for (int in_ = 0; in_ < 2; ++in_)
                acc[im][in_] = wmma16(fa[im].h, fb[in_].h, acc[im][in_]);

        if (more) {
            storeW(cur ^ 1);                // publish next buffer
            wait_async0();
        }
        __syncthreads();
    }

    #pragma unroll
    for (int in_ = 0; in_ < 2; ++in_) {
        int nc = n0 + wn * 32 + in_ * 16 + l16;
        float bv = bias[nc];
        #pragma unroll
        for (int im = 0; im < 4; ++im) {
            #pragma unroll
            for (int j = 0; j < 8; ++j) {
                int mr = m0 + wm * 64 + im * 16 + j + hi * 8;
                float v = acc[im][in_][j] + bv;
                if constexpr (OUT_F32)
                    ((float*)Cout)[(size_t)mr * N + nc] = v;
                else
                    ((_Float16*)Cout)[(size_t)mr * N + nc] = (_Float16)v;
            }
        }
    }
}

// ---------------------------------------------------------------- attention
// One block = 4 waves = one (b,h) x 64-query tile; each wave owns 16 queries.
// Double-buffered 64-key K/V tiles: K via async global->LDS DMA, V via
// reg-staged transpose. Online softmax: row-max via 4 ds_bpermute steps per
// row per tile; row-sum l accumulated by WMMA against an all-ones B matrix
// (result replicated across all lanes -> no broadcast needed anywhere).
#define KPAD 16

__global__ __launch_bounds__(128) void attn_kernel(
    const _Float16* __restrict__ qkv,   // [B*S, 3E]
    _Float16* __restrict__ aout)        // [B*S, E] (heads concatenated)
{
    __shared__ alignas(16) _Float16 Kt[2][64][64 + KPAD];  // [buf][key][d]
    __shared__ alignas(16) _Float16 Vt[2][64][64 + KPAD];  // [buf][d][key]
    __shared__ alignas(16) _Float16 Pl[4][16][64 + KPAD];  // per-wave P

    const int t    = threadIdx.x;
    const int lane = t & 31;
    const int wave = t >> 5;
    const int hi   = (lane >> 4) & 1;
    const int l16  = lane & 15;

    int bid = blockIdx.x;
    const int qt = bid & 31;  bid >>= 5;
    const int h  = bid & 15;  bid >>= 4;
    const int b  = bid;
    const int q0 = qt * 64 + wave * 16;

    uint4 vreg[4];

    auto issueK = [&](int k0, int buf) {   // 64x64: 4 b128 chunks / thread
        #pragma unroll
        for (int i = 0; i < 4; ++i) {
            int c   = t + i * 128;
            int row = c >> 3;
            int col = (c & 7) * 8;
            async_copy16(qkv + (size_t)(b * SEQ + k0 + row) * E3
                             + h * HDIM + EMB + col,
                         &Kt[buf][row][col]);
        }
    };
    auto loadV = [&](int k0) {
        #pragma unroll
        for (int i = 0; i < 4; ++i) {
            int c   = t + i * 128;
            int row = c >> 3;
            int col = (c & 7) * 8;
            vreg[i] = *(const uint4*)(qkv + (size_t)(b * SEQ + k0 + row) * E3
                                          + h * HDIM + 2 * EMB + col);
        }
    };
    auto storeV = [&](int buf) {           // scatter transposed: Vt[d][key]
        #pragma unroll
        for (int i = 0; i < 4; ++i) {
            int c   = t + i * 128;
            int row = c >> 3;
            int col = (c & 7) * 8;
            _Float16 tmp[8];
            *(uint4*)tmp = vreg[i];
            #pragma unroll
            for (int e = 0; e < 8; ++e) Vt[buf][col + e][row] = tmp[e];
        }
    };

    // Q fragments (A layout), held in registers for the whole key loop
    const _Float16* qrow = qkv + (size_t)(b * SEQ + q0 + l16) * E3 + h * HDIM;
    Frag fq0, fq1;
    fq0.u4[0] = *(const uint4*)(qrow + hi * 8);
    fq0.u4[1] = *(const uint4*)(qrow + 16 + hi * 8);
    fq1.u4[0] = *(const uint4*)(qrow + 32 + hi * 8);
    fq1.u4[1] = *(const uint4*)(qrow + 48 + hi * 8);

    v16h ones;
    #pragma unroll
    for (int i = 0; i < 16; ++i) ones[i] = (_Float16)1.0f;

    v8f acc[4] = {};
    v8f lacc   = {};
    float mrow[8];
    #pragma unroll
    for (int j = 0; j < 8; ++j) mrow[j] = -1e30f;

    // prologue: stage key-tile 0
    issueK(0, 0);
    loadV(0);
    storeV(0);
    wait_async0();
    __syncthreads();

    const int ntile = SEQ / 64;
    for (int kt = 0; kt < ntile; ++kt) {
        const int  cur  = kt & 1;
        const bool more = (kt + 1) < ntile;
        if (more) {                         // overlap next K/V fetch
            issueK((kt + 1) * 64, cur ^ 1);
            loadV((kt + 1) * 64);
        }

        // S = Q @ K^T : four 16x16 score tiles over 64 keys (K-dim = d)
        v8f s[4];
        #pragma unroll
        for (int sub = 0; sub < 4; ++sub) {
            int key = sub * 16 + l16;
            Frag fk;
            s[sub] = (v8f){};
            fk.u4[0] = *(const uint4*)&Kt[cur][key][hi * 8];
            fk.u4[1] = *(const uint4*)&Kt[cur][key][16 + hi * 8];
            s[sub] = wmma16(fq0.h, fk.h, s[sub]);
            fk.u4[0] = *(const uint4*)&Kt[cur][key][32 + hi * 8];
            fk.u4[1] = *(const uint4*)&Kt[cur][key][48 + hi * 8];
            s[sub] = wmma16(fq1.h, fk.h, s[sub]);
        }

        // online softmax per row (row j+8*hi lives in 16-lane half-groups)
        #pragma unroll
        for (int j = 0; j < 8; ++j) {
            float a0 = s[0][j] * 0.125f;   // 1/sqrt(64)
            float a1 = s[1][j] * 0.125f;
            float a2 = s[2][j] * 0.125f;
            float a3 = s[3][j] * 0.125f;
            float mx = fmaxf(fmaxf(a0, a1), fmaxf(a2, a3));
            #pragma unroll
            for (int o = 8; o >= 1; o >>= 1) mx = fmaxf(mx, __shfl_xor(mx, o, 32));
            float mnew  = fmaxf(mrow[j], mx);
            float alpha = __expf(mrow[j] - mnew);
            mrow[j] = mnew;
            float p0 = __expf(a0 - mnew);
            float p1 = __expf(a1 - mnew);
            float p2 = __expf(a2 - mnew);
            float p3 = __expf(a3 - mnew);
            lacc[j] *= alpha;
            #pragma unroll
            for (int c = 0; c < 4; ++c) acc[c][j] *= alpha;
            int prow = j + hi * 8;                  // C-layout -> row-major LDS
            Pl[wave][prow][l16]      = (_Float16)p0;
            Pl[wave][prow][16 + l16] = (_Float16)p1;
            Pl[wave][prow][32 + l16] = (_Float16)p2;
            Pl[wave][prow][48 + l16] = (_Float16)p3;
        }

        // reload P in A layout (same-wave DS ops stay in order)
        Frag fp0, fp1;
        fp0.u4[0] = *(const uint4*)&Pl[wave][l16][hi * 8];       // keys 0..31
        fp0.u4[1] = *(const uint4*)&Pl[wave][l16][16 + hi * 8];
        fp1.u4[0] = *(const uint4*)&Pl[wave][l16][32 + hi * 8];  // keys 32..63
        fp1.u4[1] = *(const uint4*)&Pl[wave][l16][48 + hi * 8];

        // l += P @ ones  (row sums, replicated into every lane)
        lacc = wmma16(fp0.h, ones, lacc);
        lacc = wmma16(fp1.h, ones, lacc);

        // O += P(16x64) @ V(64x64): 8 WMMAs over 4 d-chunks x 2 key-halves
        #pragma unroll
        for (int c = 0; c < 4; ++c) {
            int d = c * 16 + l16;
            Frag fv;
            fv.u4[0] = *(const uint4*)&Vt[cur][d][hi * 8];      // keys 0..31
            fv.u4[1] = *(const uint4*)&Vt[cur][d][16 + hi * 8];
            acc[c] = wmma16(fp0.h, fv.h, acc[c]);
            fv.u4[0] = *(const uint4*)&Vt[cur][d][32 + hi * 8]; // keys 32..63
            fv.u4[1] = *(const uint4*)&Vt[cur][d][48 + hi * 8];
            acc[c] = wmma16(fp1.h, fv.h, acc[c]);
        }

        if (more) {
            storeV(cur ^ 1);                // publish next buffer
            wait_async0();
        }
        __syncthreads();
    }

    // normalize and store (heads concatenated, f16 for the O-projection GEMM)
    #pragma unroll
    for (int c = 0; c < 4; ++c) {
        int d = c * 16 + l16;
        #pragma unroll
        for (int j = 0; j < 8; ++j) {
            int q = q0 + j + hi * 8;
            aout[(size_t)(b * SEQ + q) * EMB + h * HDIM + d] =
                (_Float16)(acc[c][j] / lacc[j]);
        }
    }
}

// ---------------------------------------------------------------- launch
extern "C" void kernel_launch(void* const* d_in, const int* in_sizes, int n_in,
                              void* d_out, int out_size, void* d_ws, size_t ws_size,
                              hipStream_t stream) {
    const float* x    = (const float*)d_in[0];
    const float* Wqkv = (const float*)d_in[1];
    const float* bqkv = (const float*)d_in[2];
    const float* Wo   = (const float*)d_in[3];
    const float* bo   = (const float*)d_in[4];

    const size_t M = (size_t)BATCH * SEQ;   // 8192 rows

    char* ws = (char*)d_ws;
    _Float16* xh    = (_Float16*)ws;  ws += M * EMB * sizeof(_Float16);
    _Float16* wqkvh = (_Float16*)ws;  ws += (size_t)EMB * E3 * sizeof(_Float16);
    _Float16* woh   = (_Float16*)ws;  ws += (size_t)EMB * EMB * sizeof(_Float16);
    _Float16* qkvh  = (_Float16*)ws;  ws += M * E3 * sizeof(_Float16);
    _Float16* aoh   = (_Float16*)ws;  ws += M * EMB * sizeof(_Float16);

    cvt_f32_to_f16<<<1024, 256, 0, stream>>>(x,    xh,    (int)(M * EMB));
    cvt_f32_to_f16<<<1024, 256, 0, stream>>>(Wqkv, wqkvh, EMB * E3);
    cvt_f32_to_f16<<<1024, 256, 0, stream>>>(Wo,   woh,   EMB * EMB);

    // QKV projection: [8192,1024] @ [1024,3072] + b  -> f16
    dim3 g1(E3 / 128, (int)(M / 128));
    gemm_f16<false><<<g1, 256, 0, stream>>>(xh, wqkvh, bqkv, (void*)qkvh,
                                            (int)M, E3, EMB);

    // attention: one block per (b, h, 64-query tile)
    attn_kernel<<<BATCH * HEADS * (SEQ / 64), 128, 0, stream>>>(qkvh, aoh);

    // output projection: [8192,1024] @ [1024,1024] + b -> f32 (d_out)
    dim3 g2(EMB / 128, (int)(M / 128));
    gemm_f16<true><<<g2, 256, 0, stream>>>(aoh, woh, bo, d_out,
                                           (int)M, EMB, EMB);
}
